// MT5Encoder_6777458393619
// MI455X (gfx1250) — compile-verified, matrix operand order checked by reading, not run on hardware
//
#include <hip/hip_runtime.h>
#include <hip/hip_bf16.h>

// ---------------- model constants ----------------
#define LNUM 4
#define D    512
#define H    8
#define DH   64
#define HD   512     // H*DH
#define DFF  1024
#define NBK  32
#define Bb   2
#define SS   2048
#define BS   (Bb*SS) // 4096
#define EPS  1e-6f

#define USE_TDM 1

typedef __bf16 bf16_t;
typedef __attribute__((ext_vector_type(16))) __bf16 v16bf;
typedef __attribute__((ext_vector_type(8)))  float  v8f;
typedef unsigned int u32x4 __attribute__((ext_vector_type(4)));
typedef int          i32x4 __attribute__((ext_vector_type(4)));
typedef int          i32x8 __attribute__((ext_vector_type(8)));

union FragU { v16bf v; uint4 u[2]; };
union Pack8 { uint4 u; bf16_t h[8]; };

#define LDSPITCH 40   // bf16 elements per LDS row (80B, 16B-aligned, padded vs 32)

#if __has_include(<hip/amd_detail/amd_gfx1250_TDM.h>)
#define TDM_6ARG 1
#endif

__device__ __forceinline__ v8f wmma_bf16f32(v16bf a, v16bf b, v8f c) {
  // D(f32) = A(bf16 16x32) * B(bf16 32x16) + C
  return __builtin_amdgcn_wmma_f32_16x16x32_bf16(false, a, false, b, (short)0, c,
                                                 false, false);
}

// Load one 16x32 bf16 fragment from an LDS tile (row-major, LDSPITCH stride).
// Lanes 0-15: row M=lane, K {0..7,16..23}; lanes 16-31: row M=lane-16,
// K {8..15,24..31}. Two ds_load_b128 per lane.
__device__ __forceinline__ v16bf frag_ld(const bf16_t* base, int rowStart, int lane) {
  int r  = rowStart + (lane & 15);
  int kh = (lane >> 4) << 3;          // 0 or 8
  const bf16_t* p = base + r * LDSPITCH + kh;
  FragU f;
  f.u[0] = *(const uint4*)(p);
  f.u[1] = *(const uint4*)(p + 16);
  return f.v;
}

// Issue a TDM 2D tile load (rows x 32 bf16, padded to LDSPITCH) into LDS.
// Descriptor bit layout per CDNA5 ISA 8.3/8.4. Wave-uniform arguments.
__device__ __forceinline__ void tdm_load_tile_a(const bf16_t* gsrc, unsigned ldsAddr,
                                                int rows, int K) {
  unsigned long long ga = (unsigned long long)(size_t)gsrc;   // byte address
  u32x4 g0;
  g0[0] = 1u;                                        // count=1, user descriptor
  g0[1] = ldsAddr;                                   // lds_addr (bytes)
  g0[2] = (unsigned)(ga & 0xFFFFFFFFu);              // global_addr[31:0]
  g0[3] = (unsigned)((ga >> 32) & 0x01FFFFFFu) | (2u << 30);  // addr[56:32], type=2
  i32x8 g1;
  // data_size=2B (code 1), pad_enable, pad_interval=16 DWORDs (code 3),
  // pad_amount=4 DWORDs (code 3) -> LDS pitch = 32 bf16 data + 8 bf16 pad = 40
  g1[0] = (1 << 16) | (1 << 20) | (3 << 22) | (3 << 25);
  g1[1] = (int)((K & 0xFFFF) << 16);                 // tensor_dim0[15:0]
  g1[2] = (int)(((unsigned)K >> 16) & 0xFFFFu) | (int)((BS & 0xFFFF) << 16); // dim0 hi | dim1 lo
  g1[3] = (int)((32u & 0xFFFFu) << 16);              // dim1 hi=0 | tile_dim0=32
  g1[4] = rows & 0xFFFF;                             // tile_dim1=rows, tile_dim2=0
  g1[5] = K;                                         // tensor_dim0_stride[31:0]
  g1[6] = 0;                                         // stride hi | dim1_stride lo
  g1[7] = 0;
  i32x4 z4 = {0, 0, 0, 0};
#if defined(TDM_6ARG)
  i32x8 z8 = {0, 0, 0, 0, 0, 0, 0, 0};
  __builtin_amdgcn_tensor_load_to_lds(g0, g1, z4, z4, z8, 0);
#else
  __builtin_amdgcn_tensor_load_to_lds(g0, g1, z4, z4, 0);
#endif
}

// T5 relative-position bucket (bidirectional, 32 buckets, max_distance 128)
__device__ __forceinline__ int rel_bucket_rel(int rel) {
  int bucket = (rel > 0) ? 16 : 0;
  int rp = rel < 0 ? -rel : rel;
  int add;
  if (rp < 8) {
    add = rp;
  } else {
    add = 8 + (int)(__logf((float)rp * 0.125f) * (8.0f / __logf(16.0f)));
    if (add > 15) add = 15;
  }
  return bucket + add;
}

// ---------------- generic bf16 GEMM: C[M,N] (+)= A[M,K] * B[K,N] ----------------
// EPI: 0 = store bf16, 1 = store f32, 2 = residual add into f32
template <int EPI>
__global__ __launch_bounds__(256) void gemm_bf16(const bf16_t* __restrict__ A,
                                                 const bf16_t* __restrict__ Bm,
                                                 void* __restrict__ C,
                                                 int M, int N, int K) {
  __shared__ bf16_t As[128 * LDSPITCH];
  __shared__ bf16_t Bs[64 * LDSPITCH];
  const int tid  = threadIdx.x;
  const int lane = tid & 31;
  const int wave = tid >> 5;
  const int wm = (wave >> 1) * 32;    // 4 waves over M (128)
  const int wn = (wave & 1) * 32;     // 2 waves over N (64)
  const int mBase = blockIdx.y * 128;
  const int nBase = blockIdx.x * 64;

  const v8f zero = {0.f, 0.f, 0.f, 0.f, 0.f, 0.f, 0.f, 0.f};
  v8f acc[2][2];
  acc[0][0] = zero; acc[0][1] = zero; acc[1][0] = zero; acc[1][1] = zero;

#if !USE_TDM
  const int ar = tid >> 1;
  const int ac = (tid & 1) << 4;
#endif
  // B tile: 32x64, 8 elems/thread, scattered transposed into Bs[n][k]
  const int bK = tid >> 3;
  const int bN = (tid & 7) << 3;
  const unsigned ldsA = (unsigned)(size_t)(void*)As;

  for (int k0 = 0; k0 < K; k0 += 32) {
#if USE_TDM
    // Wave 0 drives the Tensor Data Mover for the 128x32 A tile (padded pitch).
    if (wave == 0) {
      tdm_load_tile_a(A + (size_t)mBase * K + k0, ldsA, 128, K);
    }
#else
    const bf16_t* ag = A + (size_t)(mBase + ar) * K + k0 + ac;
    *(uint4*)(As + ar * LDSPITCH + ac)     = *(const uint4*)(ag);
    *(uint4*)(As + ar * LDSPITCH + ac + 8) = *(const uint4*)(ag + 8);
#endif
    const bf16_t* bg = Bm + (size_t)(k0 + bK) * N + nBase + bN;
    Pack8 pb;
    pb.u = *(const uint4*)(bg);
    if (k0 + 32 < K) __builtin_prefetch(bg + 32 * N, 0, 3);   // global_prefetch_b8
#pragma unroll
    for (int j = 0; j < 8; ++j) Bs[(bN + j) * LDSPITCH + bK] = pb.h[j];

#if USE_TDM
    if (wave == 0) __builtin_amdgcn_s_wait_tensorcnt(0);
#endif
    __syncthreads();
    v16bf a0 = frag_ld(As, wm, lane);
    v16bf a1 = frag_ld(As, wm + 16, lane);
    v16bf b0 = frag_ld(Bs, wn, lane);
    v16bf b1 = frag_ld(Bs, wn + 16, lane);
    acc[0][0] = wmma_bf16f32(a0, b0, acc[0][0]);
    acc[0][1] = wmma_bf16f32(a0, b1, acc[0][1]);
    acc[1][0] = wmma_bf16f32(a1, b0, acc[1][0]);
    acc[1][1] = wmma_bf16f32(a1, b1, acc[1][1]);
    __syncthreads();
  }

  const int rext = (lane >> 4) << 3;  // +8 rows for lanes 16-31
  const int cl   = lane & 15;
#pragma unroll
  for (int ms = 0; ms < 2; ++ms)
#pragma unroll
    for (int ns = 0; ns < 2; ++ns)
#pragma unroll
      for (int i = 0; i < 8; ++i) {
        int row = mBase + wm + ms * 16 + i + rext;
        int col = nBase + wn + ns * 16 + cl;
        size_t off = (size_t)row * N + col;
        float v = acc[ms][ns][i];
        if (EPI == 0)      ((bf16_t*)C)[off] = (bf16_t)v;
        else if (EPI == 1) ((float*)C)[off]  = v;
        else               ((float*)C)[off] += v;
      }
}

// ---------------- attention: scores = Q Kt + T5 bias (block-level LUT) ----------
__global__ __launch_bounds__(256) void attn_qk(const bf16_t* __restrict__ Q,
                                               const bf16_t* __restrict__ Kb,
                                               const float* __restrict__ relb,
                                               float* __restrict__ scores) {
  __shared__ bf16_t As[128 * LDSPITCH];
  __shared__ bf16_t Bs[64 * LDSPITCH];
  __shared__ float biasLut[191];        // one entry per distinct (col-row) in tile
  const int tid = threadIdx.x, lane = tid & 31, wave = tid >> 5;
  const int wm = (wave >> 1) * 32, wn = (wave & 1) * 32;
  const int mBase = blockIdx.y * 128;   // query tile
  const int nBase = blockIdx.x * 64;    // key tile
  const int bh = blockIdx.z;
  const int b = bh >> 3, h = bh & 7;
  const bf16_t* Qp = Q + (size_t)b * SS * HD + (size_t)h * DH;
  const bf16_t* Kp = Kb + (size_t)b * SS * HD + (size_t)h * DH;

  // rel = col - row ranges over [nBase-mBase-127, nBase-mBase+63]: 191 values.
  // Amortize the bucket logf over the whole 128x64 tile.
  if (tid < 191) {
    int rel = (nBase - mBase) + tid - 127;
    biasLut[tid] = relb[rel_bucket_rel(rel) * H + h];
  }

  const v8f zero = {0.f, 0.f, 0.f, 0.f, 0.f, 0.f, 0.f, 0.f};
  v8f acc[2][2];
  acc[0][0] = zero; acc[0][1] = zero; acc[1][0] = zero; acc[1][1] = zero;

  const int ar = tid >> 1, ac = (tid & 1) << 4;   // A: 128x32
  const int bn = tid >> 2, bc = (tid & 3) << 3;   // B: 64 key rows x 32 d

  for (int k0 = 0; k0 < DH; k0 += 32) {
    const bf16_t* ag = Qp + (size_t)(mBase + ar) * HD + k0 + ac;
    *(uint4*)(As + ar * LDSPITCH + ac)     = *(const uint4*)(ag);
    *(uint4*)(As + ar * LDSPITCH + ac + 8) = *(const uint4*)(ag + 8);
    const bf16_t* bg = Kp + (size_t)(nBase + bn) * HD + k0 + bc;
    *(uint4*)(Bs + bn * LDSPITCH + bc) = *(const uint4*)(bg);
    __syncthreads();
    v16bf a0 = frag_ld(As, wm, lane);
    v16bf a1 = frag_ld(As, wm + 16, lane);
    v16bf b0 = frag_ld(Bs, wn, lane);
    v16bf b1 = frag_ld(Bs, wn + 16, lane);
    acc[0][0] = wmma_bf16f32(a0, b0, acc[0][0]);
    acc[0][1] = wmma_bf16f32(a0, b1, acc[0][1]);
    acc[1][0] = wmma_bf16f32(a1, b0, acc[1][0]);
    acc[1][1] = wmma_bf16f32(a1, b1, acc[1][1]);
    __syncthreads();
  }

  const int rext = (lane >> 4) << 3;
  const int cl   = lane & 15;
#pragma unroll
  for (int ms = 0; ms < 2; ++ms)
#pragma unroll
    for (int ns = 0; ns < 2; ++ns)
#pragma unroll
      for (int i = 0; i < 8; ++i) {
        int rloc = wm + ms * 16 + i + rext;       // row - mBase
        int cloc = wn + ns * 16 + cl;             // col - nBase
        float bias = biasLut[cloc - rloc + 127];
        int row = mBase + rloc;
        int col = nBase + cloc;
        scores[((size_t)bh * SS + row) * SS + col] = acc[ms][ns][i] + bias;
      }
}

// ---------------- softmax per row, write bf16 probs in place ----------------
__global__ __launch_bounds__(256) void softmax_rows(float* __restrict__ scores) {
  float* rp = scores + (size_t)blockIdx.x * SS;
  const int tid = threadIdx.x;
  float v[8];
  float mx = -3.4e38f;
#pragma unroll
  for (int j = 0; j < 8; ++j) { v[j] = rp[tid + j * 256]; mx = fmaxf(mx, v[j]); }
  __shared__ float red[256];
  red[tid] = mx;
  __syncthreads();
  for (int off = 128; off; off >>= 1) {
    if (tid < off) red[tid] = fmaxf(red[tid], red[tid + off]);
    __syncthreads();
  }
  mx = red[0];
  __syncthreads();
  float s = 0.f;
#pragma unroll
  for (int j = 0; j < 8; ++j) { v[j] = __expf(v[j] - mx); s += v[j]; }
  red[tid] = s;
  __syncthreads();
  for (int off = 128; off; off >>= 1) {
    if (tid < off) red[tid] += red[tid + off];
    __syncthreads();
  }
  float inv = 1.f / red[0];
  bf16_t* pp = (bf16_t*)rp;   // bf16 probs overwrite first half of the fp32 row
#pragma unroll
  for (int j = 0; j < 8; ++j) pp[tid + j * 256] = (bf16_t)(v[j] * inv);
}

// ---------------- O = probs @ V ----------------
__global__ __launch_bounds__(256) void attn_av(const float* __restrict__ scoresBase,
                                               const bf16_t* __restrict__ V,
                                               bf16_t* __restrict__ O) {
  __shared__ bf16_t As[128 * LDSPITCH];
  __shared__ bf16_t Bs[64 * LDSPITCH];
  const int tid = threadIdx.x, lane = tid & 31, wave = tid >> 5;
  const int wm = (wave >> 1) * 32, wn = (wave & 1) * 32;
  const int mBase = blockIdx.y * 128;
  const int bh = blockIdx.z;
  const int b = bh >> 3, h = bh & 7;
  // probs: bf16, row q starts at float offset (bh*SS+q)*SS -> bf16 row stride 2*SS
  const bf16_t* P  = (const bf16_t*)(scoresBase + (size_t)bh * SS * SS);
  const bf16_t* Vp = V + (size_t)b * SS * HD + (size_t)h * DH;

  const v8f zero = {0.f, 0.f, 0.f, 0.f, 0.f, 0.f, 0.f, 0.f};
  v8f acc[2][2];
  acc[0][0] = zero; acc[0][1] = zero; acc[1][0] = zero; acc[1][1] = zero;

  const int ar = tid >> 1, ac = (tid & 1) << 4;   // A: 128x32 probs
  const int bK = tid >> 3, bN = (tid & 7) << 3;   // B: 32 kpos x 64 d

  for (int k0 = 0; k0 < SS; k0 += 32) {
    const bf16_t* ag = P + (size_t)(mBase + ar) * (2 * SS) + k0 + ac;
    *(uint4*)(As + ar * LDSPITCH + ac)     = *(const uint4*)(ag);
    *(uint4*)(As + ar * LDSPITCH + ac + 8) = *(const uint4*)(ag + 8);
    if (k0 + 32 < SS) __builtin_prefetch(ag + 32, 0, 3);
    Pack8 pb;
    pb.u = *(const uint4*)(Vp + (size_t)(k0 + bK) * HD + bN);
#pragma unroll
    for (int j = 0; j < 8; ++j) Bs[(bN + j) * LDSPITCH + bK] = pb.h[j];
    __syncthreads();
    v16bf a0 = frag_ld(As, wm, lane);
    v16bf a1 = frag_ld(As, wm + 16, lane);
    v16bf b0 = frag_ld(Bs, wn, lane);
    v16bf b1 = frag_ld(Bs, wn + 16, lane);
    acc[0][0] = wmma_bf16f32(a0, b0, acc[0][0]);
    acc[0][1] = wmma_bf16f32(a0, b1, acc[0][1]);
    acc[1][0] = wmma_bf16f32(a1, b0, acc[1][0]);
    acc[1][1] = wmma_bf16f32(a1, b1, acc[1][1]);
    __syncthreads();
  }

  const int rext = (lane >> 4) << 3;
  const int cl   = lane & 15;
#pragma unroll
  for (int ms = 0; ms < 2; ++ms)
#pragma unroll
    for (int ns = 0; ns < 2; ++ns)
#pragma unroll
      for (int i = 0; i < 8; ++i) {
        int row = mBase + wm + ms * 16 + i + rext;
        int col = h * DH + wn + ns * 16 + cl;
        O[(size_t)(b * SS + row) * HD + col] = (bf16_t)acc[ms][ns][i];
      }
}

// ---------------- elementwise kernels ----------------
__global__ __launch_bounds__(256) void embed_kernel(const int* __restrict__ ids,
                                                    const float* __restrict__ emb,
                                                    float* __restrict__ x) {
  int idx = blockIdx.x * 256 + threadIdx.x;   // BS*D
  int row = idx >> 9, c = idx & 511;
  x[idx] = emb[(size_t)ids[row] * D + c];
}

__global__ __launch_bounds__(256) void bias_kernel(const float* __restrict__ relb,
                                                   float* __restrict__ outb) {
  int idx = blockIdx.x * 256 + threadIdx.x;   // H*S*S
  int j = idx & (SS - 1);
  int t = idx >> 11;
  int i = t & (SS - 1);
  int h = t >> 11;
  outb[idx] = relb[rel_bucket_rel(j - i) * H + h];
}

template <int OUT_BF16>
__global__ __launch_bounds__(256) void rmsnorm_kernel(const float* __restrict__ x,
                                                      const float* __restrict__ w,
                                                      void* __restrict__ out) {
  const int row = blockIdx.x, tid = threadIdx.x;
  const float* xr = x + (size_t)row * D;
  float a = xr[tid], b = xr[tid + 256];
  __shared__ float red[256];
  red[tid] = a * a + b * b;
  __syncthreads();
  for (int off = 128; off; off >>= 1) {
    if (tid < off) red[tid] += red[tid + off];
    __syncthreads();
  }
  float inv = rsqrtf(red[0] * (1.0f / D) + EPS);
  if (OUT_BF16) {
    bf16_t* o = (bf16_t*)out + (size_t)row * D;
    o[tid]       = (bf16_t)(w[tid] * a * inv);
    o[tid + 256] = (bf16_t)(w[tid + 256] * b * inv);
  } else {
    float* o = (float*)out + (size_t)row * D;
    o[tid]       = w[tid] * a * inv;
    o[tid + 256] = w[tid + 256] * b * inv;
  }
}

__global__ __launch_bounds__(256) void gate_kernel(const float* __restrict__ g0,
                                                   const float* __restrict__ g1,
                                                   bf16_t* __restrict__ g) {
  int i = blockIdx.x * 256 + threadIdx.x;
  float xv = g0[i];
  float t = 0.5f * xv * (1.f + tanhf(0.7978845608028654f * (xv + 0.044715f * xv * xv * xv)));
  g[i] = (bf16_t)(t * g1[i]);
}

__global__ __launch_bounds__(256) void cvt_bf16_kernel(const float* __restrict__ s,
                                                       bf16_t* __restrict__ d) {
  int i = blockIdx.x * 256 + threadIdx.x;
  d[i] = (bf16_t)s[i];
}

// ---------------- orchestration ----------------
extern "C" void kernel_launch(void* const* d_in, const int* in_sizes, int n_in,
                              void* d_out, int out_size, void* d_ws, size_t ws_size,
                              hipStream_t stream) {
  (void)in_sizes; (void)n_in; (void)out_size; (void)ws_size;
  const int*   ids  = (const int*)d_in[0];
  const float* emb  = (const float*)d_in[1];
  const float* Wq   = (const float*)d_in[2];
  const float* Wk   = (const float*)d_in[3];
  const float* Wv   = (const float*)d_in[4];
  const float* Wo   = (const float*)d_in[5];
  const float* relb = (const float*)d_in[6];
  const float* wi0  = (const float*)d_in[7];
  const float* wi1  = (const float*)d_in[8];
  const float* wof  = (const float*)d_in[9];
  const float* ln0  = (const float*)d_in[10];
  const float* ln1  = (const float*)d_in[11];
  const float* lnf  = (const float*)d_in[12];

  float* out      = (float*)d_out;
  float* out_bias = out + (size_t)BS * D;   // pos_bias region of the output tuple

  char* p = (char*)d_ws;
  auto alloc = [&](size_t bytes) -> void* {
    void* r = (void*)p;
    p += (bytes + 255) & ~(size_t)255;
    return r;
  };
  const size_t WATT = (size_t)LNUM * D * HD;    // attn weight elems (per tensor)
  const size_t WFF  = (size_t)LNUM * D * DFF;   // ffn weight elems
  bf16_t* wq_b  = (bf16_t*)alloc(WATT * 2);
  bf16_t* wk_b  = (bf16_t*)alloc(WATT * 2);
  bf16_t* wv_b  = (bf16_t*)alloc(WATT * 2);
  bf16_t* wo_b  = (bf16_t*)alloc(WATT * 2);
  bf16_t* wi0_b = (bf16_t*)alloc(WFF * 2);
  bf16_t* wi1_b = (bf16_t*)alloc(WFF * 2);
  bf16_t* wof_b = (bf16_t*)alloc(WFF * 2);
  float*  x     = (float*)alloc((size_t)BS * D * 4);
  bf16_t* hbuf  = (bf16_t*)alloc((size_t)BS * D * 2);
  bf16_t* qb    = (bf16_t*)alloc((size_t)BS * HD * 2);
  bf16_t* kb    = (bf16_t*)alloc((size_t)BS * HD * 2);
  bf16_t* vb    = (bf16_t*)alloc((size_t)BS * HD * 2);
  bf16_t* ob    = (bf16_t*)alloc((size_t)BS * HD * 2);
  float*  sc    = (float*)alloc((size_t)Bb * H * SS * SS * 4);
  float*  g0    = (float*)alloc((size_t)BS * DFF * 4);
  float*  g1    = (float*)alloc((size_t)BS * DFF * 4);
  bf16_t* gb    = (bf16_t*)alloc((size_t)BS * DFF * 2);

  // weight conversion fp32 -> bf16 (once per launch)
  cvt_bf16_kernel<<<WATT / 256, 256, 0, stream>>>(Wq, wq_b);
  cvt_bf16_kernel<<<WATT / 256, 256, 0, stream>>>(Wk, wk_b);
  cvt_bf16_kernel<<<WATT / 256, 256, 0, stream>>>(Wv, wv_b);
  cvt_bf16_kernel<<<WATT / 256, 256, 0, stream>>>(Wo, wo_b);
  cvt_bf16_kernel<<<WFF / 256, 256, 0, stream>>>(wi0, wi0_b);
  cvt_bf16_kernel<<<WFF / 256, 256, 0, stream>>>(wi1, wi1_b);
  cvt_bf16_kernel<<<WFF / 256, 256, 0, stream>>>(wof, wof_b);

  embed_kernel<<<(BS * D) / 256, 256, 0, stream>>>(ids, emb, x);
  bias_kernel<<<(H * SS * SS) / 256, 256, 0, stream>>>(relb, out_bias);

  for (int l = 0; l < LNUM; ++l) {
    // --- self-attention block ---
    rmsnorm_kernel<1><<<BS, 256, 0, stream>>>(x, ln0 + (size_t)l * D, hbuf);
    gemm_bf16<0><<<dim3(HD / 64, BS / 128), 256, 0, stream>>>(hbuf, wq_b + (size_t)l * D * HD, qb, BS, HD, D);
    gemm_bf16<0><<<dim3(HD / 64, BS / 128), 256, 0, stream>>>(hbuf, wk_b + (size_t)l * D * HD, kb, BS, HD, D);
    gemm_bf16<0><<<dim3(HD / 64, BS / 128), 256, 0, stream>>>(hbuf, wv_b + (size_t)l * D * HD, vb, BS, HD, D);
    attn_qk<<<dim3(SS / 64, SS / 128, Bb * H), 256, 0, stream>>>(qb, kb, relb, sc);
    softmax_rows<<<Bb * H * SS, 256, 0, stream>>>(sc);
    attn_av<<<dim3(1, SS / 128, Bb * H), 256, 0, stream>>>(sc, vb, ob);
    gemm_bf16<2><<<dim3(D / 64, BS / 128), 256, 0, stream>>>(ob, wo_b + (size_t)l * HD * D, x, BS, D, HD);
    // --- gated-GELU FFN ---
    rmsnorm_kernel<1><<<BS, 256, 0, stream>>>(x, ln1 + (size_t)l * D, hbuf);
    gemm_bf16<1><<<dim3(DFF / 64, BS / 128), 256, 0, stream>>>(hbuf, wi0_b + (size_t)l * D * DFF, g0, BS, DFF, D);
    gemm_bf16<1><<<dim3(DFF / 64, BS / 128), 256, 0, stream>>>(hbuf, wi1_b + (size_t)l * D * DFF, g1, BS, DFF, D);
    gate_kernel<<<(BS * DFF) / 256, 256, 0, stream>>>(g0, g1, gb);
    gemm_bf16<2><<<dim3(D / 64, BS / 128), 256, 0, stream>>>(gb, wof_b + (size_t)l * DFF * D, x, BS, D, DFF);
  }
  rmsnorm_kernel<0><<<BS, 256, 0, stream>>>(x, lnf, out);
}
